// NeuroSat_80702435492251
// MI455X (gfx1250) — matrix operand order repaired and checked by reading, not compiled, not run on hardware
//
#include <hip/hip_runtime.h>

// ---------------- problem constants ----------------
static constexpr int D  = 128;
static constexpr int T  = 8;
static constexpr int NL = 8000;
static constexpr int NC = 16000;
static constexpr int CAP = 1 << 20;   // adjacency capacity per direction (E[nnz]=320K)

// ---------------- vector types ----------------
typedef __attribute__((ext_vector_type(16))) __bf16 v16bf;
typedef __attribute__((ext_vector_type(8)))  __bf16 v8bf;
typedef __attribute__((ext_vector_type(4)))  __bf16 v4bf;
typedef __attribute__((ext_vector_type(8)))  float  v8f;

__device__ __forceinline__ __bf16 f2bf(float f) {
  union { float f; unsigned u; } v; v.f = f;
  unsigned r = v.u + 0x7FFFu + ((v.u >> 16) & 1u);   // round-to-nearest-even
  unsigned short h = (unsigned short)(r >> 16);
  union { unsigned short s; __bf16 b; } o; o.s = h; return o.b;
}

// ---------------- f32 -> bf16 bulk conversion (vectorized, n % 4 == 0) -------
__global__ __launch_bounds__(256) void cvt_bf16(const float* __restrict__ src,
                                                __bf16* __restrict__ dst, int n4) {
  int i = blockIdx.x * blockDim.x + threadIdx.x;
  if (i >= n4) return;
  float4 f = ((const float4*)src)[i];
  v4bf o;
  o[0] = f2bf(f.x); o[1] = f2bf(f.y); o[2] = f2bf(f.z); o[3] = f2bf(f.w);
  ((v4bf*)dst)[i] = o;
}

// ---------------- CSR build: count pass ----------------
__global__ __launch_bounds__(128) void build_count(const float* __restrict__ M,
                                                   int* __restrict__ lit_cnt,
                                                   int* __restrict__ cl_cnt) {
  int wave = blockIdx.x * (blockDim.x >> 5) + (threadIdx.x >> 5);
  int lane = threadIdx.x & 31;
  if (wave >= NL) return;
  const float* row = M + (size_t)wave * NC;
  int cnt = 0;
  for (int c0 = 0; c0 < NC; c0 += 32) {
    float v = row[c0 + lane];
    unsigned long long mask = __ballot(v != 0.0f);
    cnt += __popcll(mask);
    if (v != 0.0f) atomicAdd(&cl_cnt[c0 + lane], 1);
  }
  if (lane == 0) lit_cnt[wave] = cnt;
}

// ---------------- exclusive scan (single block, LDS) ----------------
__global__ void scan_excl(const int* __restrict__ cnt, int* __restrict__ ptr, int n) {
  __shared__ int sh[1024];
  int tid = threadIdx.x;
  int carry = 0;
  for (int base = 0; base < n; base += 1024) {
    int i = base + tid;
    int v = (i < n) ? cnt[i] : 0;
    sh[tid] = v; __syncthreads();
    for (int off = 1; off < 1024; off <<= 1) {
      int t = (tid >= off) ? sh[tid - off] : 0;
      __syncthreads();
      sh[tid] += t;
      __syncthreads();
    }
    if (i < n) ptr[i + 1] = carry + sh[tid];
    int tot = sh[1023];
    __syncthreads();
    carry += tot;
  }
  if (tid == 0) ptr[0] = 0;
}

// ---------------- CSR build: fill pass ----------------
__global__ __launch_bounds__(128) void build_fill(const float* __restrict__ M,
                                                  const int* __restrict__ lit_ptr,
                                                  const int* __restrict__ cl_ptr,
                                                  int* __restrict__ cl_cur,
                                                  int* __restrict__ lit_idx,
                                                  int* __restrict__ cl_idx) {
  int wave = blockIdx.x * (blockDim.x >> 5) + (threadIdx.x >> 5);
  int lane = threadIdx.x & 31;
  if (wave >= NL) return;
  const float* row = M + (size_t)wave * NC;
  int base = lit_ptr[wave];
  for (int c0 = 0; c0 < NC; c0 += 32) {
    float v = row[c0 + lane];
    unsigned long long mask = __ballot(v != 0.0f);
    int before = __popcll(mask & ((1ull << lane) - 1ull));
    if (v != 0.0f) {
      int p = base + before;
      if (p < CAP) lit_idx[p] = c0 + lane;
      int c = c0 + lane;
      int q = cl_ptr[c] + atomicAdd(&cl_cur[c], 1);
      if (q < CAP) cl_idx[q] = wave;
    }
    base += __popcll(mask);
  }
}

// insertion sort each clause's literal list -> deterministic replay order
__global__ void sort_lists(const int* __restrict__ ptr, int* __restrict__ idx, int n) {
  int r = blockIdx.x * blockDim.x + threadIdx.x;
  if (r >= n) return;
  int b = ptr[r], e = ptr[r + 1];
  if (b > CAP) b = CAP;
  if (e > CAP) e = CAP;
  for (int i = b + 1; i < e; ++i) {
    int key = idx[i];
    int j = i - 1;
    while (j >= b && idx[j] > key) { idx[j + 1] = idx[j]; --j; }
    idx[j + 1] = key;
  }
}

// ---------------- fused multi-term bf16 GEMM ---------------------------------
// Y(R x O, f32) = sum_t X_t(R x K_t, bf16) @ W_t(O x K_t, bf16)^T + bias0 + bias1
// Each wave computes a 16x64 strip: 1 A-fragment load feeds 4 WMMAs.
__device__ __forceinline__ void wmma_term4(v8f acc[4], const __bf16* X, int ldx,
                                           const __bf16* W, int ldw, int K,
                                           int row0, int col0, int half, int mn) {
  if (!X) return;
  const __bf16* xrow = X + (size_t)(row0 + mn) * ldx;   // A: M = lane&15
  const __bf16* wrow = W + (size_t)(col0 + mn) * ldw;   // B: N = lane&15 (+16n)
  for (int k0 = 0; k0 < K; k0 += 32) {
    union { v16bf v; v8bf h[2]; } a;
    a.h[0] = *(const v8bf*)(xrow + k0 + half * 8);        // K 0-7  / 8-15
    a.h[1] = *(const v8bf*)(xrow + k0 + 16 + half * 8);   // K 16-23 / 24-31
    #pragma unroll
    for (int n = 0; n < 4; ++n) {
      union { v16bf v; v8bf h[2]; } bu;
      bu.h[0] = *(const v8bf*)(wrow + (size_t)(n * 16) * ldw + k0 + half * 8);
      bu.h[1] = *(const v8bf*)(wrow + (size_t)(n * 16) * ldw + k0 + 16 + half * 8);
      v16bf b = bu.v;
      acc[n] = __builtin_amdgcn_wmma_f32_16x16x32_bf16(false, a.v, false, b,
                                                       (short)0, acc[n], false, false);
    }
  }
}

__global__ __launch_bounds__(128) void gemm_wmma(
    const __bf16* X0, int ldx0, const __bf16* W0, int ldw0, int K0,
    const __bf16* X1, int ldx1, const __bf16* W1, int ldw1, int K1,
    const __bf16* X2, int ldx2, const __bf16* W2, int ldw2, int K2,
    const float* bias0, const float* bias1,
    float* __restrict__ Y, int ldy) {
  int wave = threadIdx.x >> 5, lane = threadIdx.x & 31;
  int row0 = blockIdx.y * 64 + wave * 16;
  int col0 = blockIdx.x * 64;
  int half = lane >> 4, mn = lane & 15;
  v8f acc[4] = {{}, {}, {}, {}};
  wmma_term4(acc, X0, ldx0, W0, ldw0, K0, row0, col0, half, mn);
  wmma_term4(acc, X1, ldx1, W1, ldw1, K1, row0, col0, half, mn);
  wmma_term4(acc, X2, ldx2, W2, ldw2, K2, row0, col0, half, mn);
  #pragma unroll
  for (int n = 0; n < 4; ++n) {
    int colg = col0 + n * 16 + mn;
    float bs = 0.0f;
    if (bias0) bs += bias0[colg];
    if (bias1) bs += bias1[colg];
    #pragma unroll
    for (int r = 0; r < 8; ++r) {             // D layout: VGPR r -> row r (+8 hi half)
      int m = r + half * 8;
      Y[(size_t)(row0 + m) * ldy + colg] = acc[n][r] + bs;
    }
  }
}

// ---------------- SpMM gather, bf16 output: Y[r,:] = sum X[idx,:] ------------
__global__ __launch_bounds__(128) void spmm_rows(const float* __restrict__ X,
                                                 const int* __restrict__ ptr,
                                                 const int* __restrict__ idx,
                                                 __bf16* __restrict__ Y) {
  int r = blockIdx.x, d = threadIdx.x;
  int b = ptr[r], e = ptr[r + 1];
  if (b > CAP) b = CAP;
  if (e > CAP) e = CAP;
  float acc = 0.0f;
  for (int i = b; i < e; ++i) {
    int c = idx[i];
    if (i + 1 < e) __builtin_prefetch(&X[(size_t)idx[i + 1] * D + d], 0, 1);
    acc += X[(size_t)c * D + d];
  }
  Y[(size_t)r * D + d] = f2bf(acc);
}

// ---------------- fused LSTM elementwise -------------------------------------
// h and c inputs alias (reference passes the same buffer for both). Emits bf16
// copies of h (matmul operand) and c (next round's h-operand) alongside f32 c.
__device__ __forceinline__ float sigm(float x) { return 1.0f / (1.0f + __expf(-x)); }

__global__ __launch_bounds__(128) void lstm_ew(const float* __restrict__ gates,
                                               float* __restrict__ h_f32,   // may be null
                                               __bf16* __restrict__ h_bf,
                                               float* __restrict__ c_io,
                                               __bf16* __restrict__ c_bf) {
  int n = blockIdx.x, d = threadIdx.x;
  const float* g = gates + (size_t)n * (4 * D);
  float i = sigm(g[d]);
  float f = sigm(g[D + d]);
  float z = tanhf(g[2 * D + d]);
  float o = sigm(g[3 * D + d]);
  size_t p = (size_t)n * D + d;
  float c_new = f * c_io[p] + i * z;
  c_io[p] = c_new;
  c_bf[p] = f2bf(c_new);
  float h = o * tanhf(c_new);
  if (h_f32) h_f32[p] = h;
  h_bf[p] = f2bf(h);
}

// ---------------- host-side orchestration ----------------
extern "C" void kernel_launch(void* const* d_in, const int* in_sizes, int n_in,
                              void* d_out, int out_size, void* d_ws, size_t ws_size,
                              hipStream_t stream) {
  (void)in_sizes; (void)n_in; (void)out_size; (void)ws_size;

  const float* L_state  = (const float*)d_in[0];
  // d_in[1] = C_state (unused), d_in[5] = n_vars (flip is identity)
  const float* hidden_L = (const float*)d_in[2];
  const float* hidden_C = (const float*)d_in[3];
  const float* M        = (const float*)d_in[4];
  const float* W_lc  = (const float*)d_in[6];
  const float* b_lc  = (const float*)d_in[7];
  const float* W_cl  = (const float*)d_in[8];
  const float* b_cl  = (const float*)d_in[9];
  const float* Wih_C = (const float*)d_in[10];
  const float* Whh_C = (const float*)d_in[11];
  const float* bih_C = (const float*)d_in[12];
  const float* bhh_C = (const float*)d_in[13];
  const float* Wih_L = (const float*)d_in[14];
  const float* Whh_L = (const float*)d_in[15];
  const float* bih_L = (const float*)d_in[16];
  const float* bhh_L = (const float*)d_in[17];

  char* base = (char*)d_ws;
  size_t off = 0;
  auto carve = [&](size_t bytes) -> void* {
    void* p = base + off;
    off = (off + bytes + 255) & ~(size_t)255;
    return p;
  };
  // f32 state / intermediates
  float* L      = (float*)carve((size_t)NL * D * 4);
  float* hL     = (float*)carve((size_t)NL * D * 4);
  float* hC     = (float*)carve((size_t)NC * D * 4);
  float* LC_pre = (float*)carve((size_t)NL * D * 4);
  float* CL_pre = (float*)carve((size_t)NC * D * 4);
  float* gates  = (float*)carve((size_t)NC * 4 * D * 4);   // shared by both LSTMs
  // bf16 matmul operands
  __bf16* Lh    = (__bf16*)carve((size_t)NL * D * 2);
  __bf16* hLh   = (__bf16*)carve((size_t)NL * D * 2);
  __bf16* hCh   = (__bf16*)carve((size_t)NC * D * 2);
  __bf16* Csh   = (__bf16*)carve((size_t)NC * D * 2);
  __bf16* LCmh  = (__bf16*)carve((size_t)NC * D * 2);
  __bf16* CLmh  = (__bf16*)carve((size_t)NL * D * 2);
  __bf16* W_lc_h  = (__bf16*)carve((size_t)D * D * 2);
  __bf16* W_cl_h  = (__bf16*)carve((size_t)D * D * 2);
  __bf16* Wih_C_h = (__bf16*)carve((size_t)4 * D * D * 2);
  __bf16* Whh_C_h = (__bf16*)carve((size_t)4 * D * D * 2);
  __bf16* Wih_L_h = (__bf16*)carve((size_t)4 * D * 2 * D * 2);
  __bf16* Whh_L_h = (__bf16*)carve((size_t)4 * D * D * 2);
  // sparse structure
  int* lit_ptr = (int*)carve((size_t)(NL + 1) * 4);
  int* cl_ptr  = (int*)carve((size_t)(NC + 1) * 4);
  int* lit_cnt = (int*)carve((size_t)NL * 4);
  int* cl_cnt  = (int*)carve((size_t)NC * 4);              // reused as fill cursor
  int* lit_idx = (int*)carve((size_t)CAP * 4);
  int* cl_idx  = (int*)carve((size_t)CAP * 4);

  auto cvt = [&](const float* s, __bf16* d, int n) {
    int n4 = n / 4;
    cvt_bf16<<<(n4 + 255) / 256, 256, 0, stream>>>(s, d, n4);
  };

  // init f32 state (do not mutate inputs)
  hipMemcpyAsync(L,  L_state,  (size_t)NL * D * 4, hipMemcpyDeviceToDevice, stream);
  hipMemcpyAsync(hL, hidden_L, (size_t)NL * D * 4, hipMemcpyDeviceToDevice, stream);
  hipMemcpyAsync(hC, hidden_C, (size_t)NC * D * 4, hipMemcpyDeviceToDevice, stream);
  // one-time bf16 conversions: weights + initial states
  cvt(W_lc,  W_lc_h,  D * D);
  cvt(W_cl,  W_cl_h,  D * D);
  cvt(Wih_C, Wih_C_h, 4 * D * D);
  cvt(Whh_C, Whh_C_h, 4 * D * D);
  cvt(Wih_L, Wih_L_h, 4 * D * 2 * D);
  cvt(Whh_L, Whh_L_h, 4 * D * D);
  cvt(L_state,  Lh,  NL * D);
  cvt(hidden_L, hLh, NL * D);
  cvt(hidden_C, hCh, NC * D);

  // ---- build sparse adjacency from dense M (once per launch, in-graph) ----
  hipMemsetAsync(lit_cnt, 0, (size_t)NL * 4, stream);
  hipMemsetAsync(cl_cnt,  0, (size_t)NC * 4, stream);
  build_count<<<NL / 4, 128, 0, stream>>>(M, lit_cnt, cl_cnt);
  scan_excl<<<1, 1024, 0, stream>>>(lit_cnt, lit_ptr, NL);
  scan_excl<<<1, 1024, 0, stream>>>(cl_cnt,  cl_ptr,  NC);
  hipMemsetAsync(cl_cnt, 0, (size_t)NC * 4, stream);
  build_fill<<<NL / 4, 128, 0, stream>>>(M, lit_ptr, cl_ptr, cl_cnt, lit_idx, cl_idx);
  sort_lists<<<(NC + 255) / 256, 256, 0, stream>>>(cl_ptr, cl_idx, NC);

  // ---- T rounds of message passing ----
  for (int t = 0; t < T; ++t) {
    // LC_pre = L @ W_lc^T + b_lc                     (8000 x 128, f32)
    gemm_wmma<<<dim3(D / 64, NL / 64), 128, 0, stream>>>(
        Lh, D, W_lc_h, D, D,  nullptr, 0, nullptr, 0, 0,  nullptr, 0, nullptr, 0, 0,
        b_lc, nullptr, LC_pre, D);
    // LC_msgs = M^T @ LC_pre  (per-clause gather, bf16 out)
    spmm_rows<<<NC, D, 0, stream>>>(LC_pre, cl_ptr, cl_idx, LCmh);
    // clause gates = LC_msgs @ Wih_C^T + hC @ Whh_C^T + biases
    gemm_wmma<<<dim3(4 * D / 64, NC / 64), 128, 0, stream>>>(
        LCmh, D, Wih_C_h, D, D,  hCh, D, Whh_C_h, D, D,  nullptr, 0, nullptr, 0, 0,
        bih_C, bhh_C, gates, 4 * D);
    lstm_ew<<<NC, D, 0, stream>>>(gates, nullptr, Csh, hC, hCh);
    // CL_pre = Cs @ W_cl^T + b_cl                    (16000 x 128, f32)
    gemm_wmma<<<dim3(D / 64, NC / 64), 128, 0, stream>>>(
        Csh, D, W_cl_h, D, D,  nullptr, 0, nullptr, 0, 0,  nullptr, 0, nullptr, 0, 0,
        b_cl, nullptr, CL_pre, D);
    // CL_msgs = M @ CL_pre (per-literal gather, bf16 out)
    spmm_rows<<<NL, D, 0, stream>>>(CL_pre, lit_ptr, lit_idx, CLmh);
    // literal gates = [CL_msgs, L] @ Wih_L^T + hL @ Whh_L^T + biases
    gemm_wmma<<<dim3(4 * D / 64, NL / 64), 128, 0, stream>>>(
        CLmh, D, Wih_L_h,     2 * D, D,
        Lh,   D, Wih_L_h + D, 2 * D, D,
        hLh,  D, Whh_L_h,     D,     D,
        bih_L, bhh_L, gates, 4 * D);
    lstm_ew<<<NL, D, 0, stream>>>(gates, L, Lh, hL, hLh);   // L=h_new, hL=c_new
  }

  hipMemcpyAsync(d_out, L, (size_t)NL * D * 4, hipMemcpyDeviceToDevice, stream);
}
